// ICLLoss_16827681866135
// MI455X (gfx1250) — matrix operand order repaired and checked by reading, not compiled, not run on hardware
//
#include <hip/hip_runtime.h>
#include <hip/hip_bf16.h>

typedef _Float16 half_t;
typedef __attribute__((ext_vector_type(16))) _Float16 v16h;
typedef __attribute__((ext_vector_type(8)))  _Float16 v8h;
typedef __attribute__((ext_vector_type(8)))  float    v8f;
typedef __attribute__((ext_vector_type(4)))  float    v4f;

#define B_   32
#define P_   1024
#define O_   128
#define D_   512
#define TINV 10.0f      // 1/TEMP
#define EPS_ 1e-8f

// ---------------------------------------------------------------------------
// Kernel 1: L2-normalize each D=512 row of patch/obj features, emit f16.
// One wave (32 lanes) per row; each lane owns 16 consecutive floats.
// ---------------------------------------------------------------------------
__global__ void __launch_bounds__(256)
normalize_f16_kernel(const float* __restrict__ pf,
                     const float* __restrict__ of,
                     half_t* __restrict__ pn16,
                     half_t* __restrict__ on16) {
    const int wave = threadIdx.x >> 5;
    const int lane = threadIdx.x & 31;
    const long long row = (long long)blockIdx.x * 8 + wave;
    const long long nrowsP = (long long)B_ * P_;
    const long long nrows  = nrowsP + (long long)B_ * O_;
    if (row >= nrows) return;

    const float* src;
    half_t* dst;
    if (row < nrowsP) { src = pf + row * D_;            dst = pn16 + row * D_; }
    else              { src = of + (row - nrowsP) * D_; dst = on16 + (row - nrowsP) * D_; }

    v4f x[4];
    const v4f* s4 = (const v4f*)(src + lane * 16);
    float ss = 0.f;
#pragma unroll
    for (int i = 0; i < 4; ++i) {
        x[i] = s4[i];
        ss += x[i][0]*x[i][0] + x[i][1]*x[i][1] + x[i][2]*x[i][2] + x[i][3]*x[i][3];
    }
#pragma unroll
    for (int m = 16; m >= 1; m >>= 1) ss += __shfl_xor(ss, m, 32);
    const float inv = 1.0f / fmaxf(sqrtf(ss), 1e-12f);

    v8h h0, h1;
#pragma unroll
    for (int j = 0; j < 8; ++j) {
        h0[j] = (half_t)(x[j >> 2][j & 3] * inv);
        h1[j] = (half_t)(x[(j + 8) >> 2][(j + 8) & 3] * inv);
    }
    v8h* d8 = (v8h*)(dst + lane * 16);
    d8[0] = h0; d8[1] = h1;
}

// ---------------------------------------------------------------------------
// Kernel 2: fused masked-exp-sum GEMM.
// 2 waves per block; each wave owns a 16-row A tile (staged in LDS) and
// iterates 64 pp q-tiles + 8 po o-tiles, 16x v_wmma_f32_16x16x32_f16 per tile.
// C layout (16x16 f32): lane = {hi=lane>>4, n=lane&15}; vgpr r -> row m=r+8*hi.
// ---------------------------------------------------------------------------
__global__ void __launch_bounds__(64)
icl_tile_kernel(const half_t* __restrict__ pn16,
                const half_t* __restrict__ on16,
                const float*  __restrict__ e1i,
                const float*  __restrict__ e1j,
                const float*  __restrict__ e2j,
                float* __restrict__ lossArr,
                float* __restrict__ validArr,
                float* __restrict__ labArr) {
    __shared__ half_t atile[2 * 16 * D_];     // 32 KB, one 16x512 tile per wave

    const int lane = threadIdx.x & 31;
    const int wave = threadIdx.x >> 5;
    const int bb   = blockIdx.y;
    const int p0   = (blockIdx.x * 2 + wave) * 16;
    const int n    = lane & 15;               // output column within tile
    const int hi   = lane >> 4;               // lane-half selector

    half_t* lds = atile + wave * (16 * D_);

    // ---- stage A tile (16 rows x 512 K, contiguous in memory) into LDS ----
    {
        const v8h* g = (const v8h*)(pn16 + ((size_t)bb * P_ + p0) * D_);
        v8h* l = (v8h*)lds;
        for (int i = lane; i < (16 * D_) / 8; i += 32) l[i] = g[i];
    }
    __syncthreads();

    // A fragment (ISA 16-bit A 16x32 layout): lane holds row m=lane&15;
    // vgprs 0..3 = K k0+hi*8 .. +7 ; vgprs 4..7 = K k0+16+hi*8 .. +7
    auto load_a = [&](int kc) {
        const int k0 = kc * 32;
        const v8h q0 = *(const v8h*)&lds[(lane & 15) * D_ + k0 + hi * 8];
        const v8h q1 = *(const v8h*)&lds[(lane & 15) * D_ + k0 + 16 + hi * 8];
        v16h a;
#pragma unroll
        for (int i = 0; i < 8; ++i) { a[i] = q0[i]; a[i + 8] = q1[i]; }
        return a;
    };
    // B fragment: lane = column n (+half); K k0+hi*8..+7 then k0+16+hi*8..+7
    auto load_b = [&](const half_t* Bbase, int col, int kc) {
        const v8h* q = (const v8h*)(Bbase + (size_t)col * D_ + kc * 32);
        const v8h q0 = q[hi], q1 = q[2 + hi];
        v16h b;
#pragma unroll
        for (int i = 0; i < 8; ++i) { b[i] = q0[i]; b[i + 8] = q1[i]; }
        return b;
    };

    // ================= patch-patch part: sum_q e1j * exp(sim/T) ============
    float accE1j[8];
#pragma unroll
    for (int r = 0; r < 8; ++r) accE1j[r] = 0.f;

    const half_t* Bpp  = pn16 + (size_t)bb * P_ * D_;
    const float*  e1jB = e1j + ((size_t)bb * P_ + p0) * P_;

    for (int qt = 0; qt < P_ / 16; ++qt) {
        const int q0 = qt * 16;
        if (qt + 1 < P_ / 16)   // hint next B tile row into cache
            __builtin_prefetch(Bpp + (size_t)(q0 + 16 + n) * D_, 0, 1);
        v8f c = {};
#pragma unroll
        for (int kc = 0; kc < 16; ++kc) {
            v16h a = load_a(kc);
            v16h b = load_b(Bpp, q0 + n, kc);
            c = __builtin_amdgcn_wmma_f32_16x16x32_f16(
                    false, a, false, b, (short)0, c, false, false);
        }
#pragma unroll
        for (int r = 0; r < 8; ++r) {
            const int m = r + hi * 8;
            const float e   = __expf(c[r] * TINV);
            const float msk = e1jB[(size_t)m * P_ + q0 + n];
            accE1j[r] += msk * e;
        }
    }

    // ================= patch-object part ===================================
    float accE2j[8], accDel[8], accVal[8], maxv[8], lblv[8];
    int   maxo[8];
#pragma unroll
    for (int r = 0; r < 8; ++r) {
        accE2j[r] = 0.f; accDel[r] = 0.f; accVal[r] = 0.f;
        maxv[r] = -1e30f; maxo[r] = 0; lblv[r] = 0.f;
    }

    const half_t* Bpo  = on16 + (size_t)bb * O_ * D_;
    const float*  e2jB = e2j + ((size_t)bb * P_ + p0) * O_;
    const float*  e1iB = e1i + ((size_t)bb * P_ + p0) * O_;

    for (int ot = 0; ot < O_ / 16; ++ot) {
        const int o0 = ot * 16;
        v8f c = {};
#pragma unroll
        for (int kc = 0; kc < 16; ++kc) {
            v16h a = load_a(kc);
            v16h b = load_b(Bpo, o0 + n, kc);
            c = __builtin_amdgcn_wmma_f32_16x16x32_f16(
                    false, a, false, b, (short)0, c, false, false);
        }
#pragma unroll
        for (int r = 0; r < 8; ++r) {
            const int m = r + hi * 8;
            const float s  = c[r];
            const float e  = __expf(s * TINV);
            const size_t mi = (size_t)m * O_ + o0 + n;
            const float m2 = e2jB[mi];
            const float m1 = e1iB[mi];
            accE2j[r] += m2 * e;
            accDel[r] += m1 * e;
            accVal[r] += m1;
            if (s > maxv[r]) { maxv[r] = s; maxo[r] = o0 + n; lblv[r] = m1; }
        }
    }

    // ============ reduce across the 16 N-lanes of each lane-half ===========
#pragma unroll
    for (int mk = 8; mk >= 1; mk >>= 1) {
#pragma unroll
        for (int r = 0; r < 8; ++r) {
            accE1j[r] += __shfl_xor(accE1j[r], mk, 32);
            accE2j[r] += __shfl_xor(accE2j[r], mk, 32);
            accDel[r] += __shfl_xor(accDel[r], mk, 32);
            accVal[r] += __shfl_xor(accVal[r], mk, 32);
            const float ov = __shfl_xor(maxv[r], mk, 32);
            const int   oo = __shfl_xor(maxo[r], mk, 32);
            const float ol = __shfl_xor(lblv[r], mk, 32);
            if (ov > maxv[r] || (ov == maxv[r] && oo < maxo[r])) {
                maxv[r] = ov; maxo[r] = oo; lblv[r] = ol;
            }
        }
    }

    // lanes 0 / 16 hold final rows {0..7} / {8..15}
    if (n == 0) {
#pragma unroll
        for (int r = 0; r < 8; ++r) {
            const int m = r + hi * 8;
            const size_t idx = (size_t)bb * P_ + p0 + m;
            const float delta = accDel[r];
            const float ratio = delta / (delta + accE1j[r] + accE2j[r] + EPS_);
            const float loss  = -__logf(ratio + EPS_);
            const float valid = accVal[r] > 0.f ? 1.f : 0.f;
            lossArr[idx]  = loss * valid;
            validArr[idx] = valid;
            labArr[idx]   = lblv[r] * valid;
        }
    }
}

// ---------------------------------------------------------------------------
// Kernel 3: final tree reduction -> loss, matched_success_ratio
// ---------------------------------------------------------------------------
__global__ void __launch_bounds__(1024)
final_reduce_kernel(const float* __restrict__ lossArr,
                    const float* __restrict__ validArr,
                    const float* __restrict__ labArr,
                    float* __restrict__ out) {
    __shared__ float sl[1024], sv[1024], sb[1024];
    float a = 0.f, b = 0.f, c = 0.f;
    for (int i = threadIdx.x; i < B_ * P_; i += 1024) {
        a += lossArr[i]; b += validArr[i]; c += labArr[i];
    }
    sl[threadIdx.x] = a; sv[threadIdx.x] = b; sb[threadIdx.x] = c;
    __syncthreads();
    for (int s = 512; s > 0; s >>= 1) {
        if (threadIdx.x < s) {
            sl[threadIdx.x] += sl[threadIdx.x + s];
            sv[threadIdx.x] += sv[threadIdx.x + s];
            sb[threadIdx.x] += sb[threadIdx.x + s];
        }
        __syncthreads();
    }
    if (threadIdx.x == 0) {
        out[0] = sl[0] / sv[0];
        out[1] = sb[0] / sv[0];
    }
}

// ---------------------------------------------------------------------------
extern "C" void kernel_launch(void* const* d_in, const int* in_sizes, int n_in,
                              void* d_out, int out_size, void* d_ws, size_t ws_size,
                              hipStream_t stream) {
    (void)in_sizes; (void)n_in; (void)out_size; (void)ws_size;
    const float* pf  = (const float*)d_in[0];   // [B,Ph,Pw,D] = [B,P,D]
    const float* of  = (const float*)d_in[1];   // [B,O,D]
    const float* e1i = (const float*)d_in[2];   // [B,P,O]
    const float* e1j = (const float*)d_in[3];   // [B,P,P]
    const float* e2j = (const float*)d_in[4];   // [B,P,O]
    float* out = (float*)d_out;                 // {loss, matched_success_ratio}

    char* ws = (char*)d_ws;
    const size_t pnBytes = (size_t)B_ * P_ * D_ * sizeof(half_t);   // 32 MB
    const size_t onBytes = (size_t)B_ * O_ * D_ * sizeof(half_t);   //  4 MB
    half_t* pn16    = (half_t*)ws;
    half_t* on16    = (half_t*)(ws + pnBytes);
    float* lossArr  = (float*)(ws + pnBytes + onBytes);
    float* validArr = lossArr + B_ * P_;
    float* labArr   = validArr + B_ * P_;

    const int nrows = B_ * P_ + B_ * O_;
    normalize_f16_kernel<<<(nrows + 7) / 8, 256, 0, stream>>>(pf, of, pn16, on16);

    dim3 grid(P_ / 32, B_);   // 2 waves per block, each owns a 16-row tile
    icl_tile_kernel<<<grid, 64, 0, stream>>>(pn16, on16, e1i, e1j, e2j,
                                             lossArr, validArr, labArr);

    final_reduce_kernel<<<1, 1024, 0, stream>>>(lossArr, validArr, labArr, out);
}
